// WindowAttention_23364622090751
// MI455X (gfx1250) — compile-verified, matrix operand order checked by reading, not run on hardware
//
#include <hip/hip_runtime.h>
#include <hip/hip_bf16.h>

typedef __attribute__((ext_vector_type(16))) __bf16 bf16x16;
typedef __attribute__((ext_vector_type(8)))  float  f32x8;

// ---- geometry ----
// B=16, C=256, H=W=64, NH=8, WS=8, D=32, HH=WW=8
#define NBATCH 16
#define CCH    256
#define HW     4096
#define NHEAD  8
#define DHEAD  32
#define NWIN   64
#define NPOS   64

// one q/k/v part in bf16 elements: 16*8*64*64*32
static constexpr size_t PART = (size_t)NBATCH * NHEAD * NWIN * NPOS * DHEAD; // 16,777,216

static __device__ inline f32x8 wmma_bf16(bf16x16 a, bf16x16 b, f32x8 c) {
    return __builtin_amdgcn_wmma_f32_16x16x32_bf16(
        /*neg_a=*/false, a, /*neg_b=*/false, b,
        /*c_mod=*/(short)0, c, /*reuse_a=*/false, /*reuse_b=*/false);
}

// A/B fragment gather from a row-major bf16 buffer; `p` must already point at
// row_base + kbase. ISA layout: lane holds K = {k0..k0+7, k0+16..k0+23},
// k0 = 8 for lanes 16..31. Each half-run is 8 contiguous bf16 = 16B.
static __device__ inline bf16x16 frag_bf16(const __bf16* p, int lane) {
    const int k0 = (lane & 16) ? 8 : 0;
    bf16x16 f;
#pragma unroll
    for (int e = 0; e < 8; ++e) f[e]     = p[k0 + e];
#pragma unroll
    for (int e = 0; e < 8; ++e) f[8 + e] = p[k0 + 16 + e];
    return f;
}

// Same gather from contiguous fp32 (weight rows), converting to bf16.
static __device__ inline bf16x16 frag_f32_row(const float* p, int lane) {
    const int k0 = (lane & 16) ? 8 : 0;
    bf16x16 f;
#pragma unroll
    for (int e = 0; e < 8; ++e) f[e]     = (__bf16)p[k0 + e];
#pragma unroll
    for (int e = 0; e < 8; ++e) f[8 + e] = (__bf16)p[k0 + 16 + e];
    return f;
}

// Strided column gather (B operand from a [K][N] row-major tensor), fp32 source.
static __device__ inline bf16x16 frag_f32_col(const float* colbase, int stride,
                                              int kbase, int lane) {
    const int k0 = kbase + ((lane & 16) ? 8 : 0);
    bf16x16 f;
#pragma unroll
    for (int e = 0; e < 8; ++e) f[e]     = (__bf16)colbase[(size_t)(k0 + e) * stride];
#pragma unroll
    for (int e = 0; e < 8; ++e) f[8 + e] = (__bf16)colbase[(size_t)(k0 + 16 + e) * stride];
    return f;
}

// Strided column gather, bf16 source.
static __device__ inline bf16x16 frag_bf16_col(const __bf16* colbase, int stride,
                                               int kbase, int lane) {
    const int k0 = kbase + ((lane & 16) ? 8 : 0);
    bf16x16 f;
#pragma unroll
    for (int e = 0; e < 8; ++e) f[e]     = colbase[(size_t)(k0 + e) * stride];
#pragma unroll
    for (int e = 0; e < 8; ++e) f[8 + e] = colbase[(size_t)(k0 + 16 + e) * stride];
    return f;
}

// ---------------------------------------------------------------------------
// Kernel 1: qkv = w_qkv @ x + b, written as bf16 directly in windowed layout
//   qkvws[part][b][head][win][p][d],  win = hh*8+ww, p = ws1*8+ws2,
//   pixel (h,w) = (ws1*8+hh, ws2*8+ww)   [dilated windows, per reference]
// Each wave computes a 64x64 output block (4x4 WMMA tiles) for 4x L2 reuse.
// grid = (12 Mblocks, 64 Nblocks, 16 batches), 1 wave/block.
// ---------------------------------------------------------------------------
__global__ __launch_bounds__(32) void qkv_gemm(const float* __restrict__ x,
                                               const float* __restrict__ w,
                                               const float* __restrict__ bias,
                                               __bf16* __restrict__ qkvws) {
    const int lane = threadIdx.x;
    const int l15  = lane & 15;
    const int hi   = (lane >> 4) & 1;
    const int tmb  = blockIdx.x;   // 0..11  (768/64)
    const int tnb  = blockIdx.y;   // 0..63  (4096/64)
    const int b    = blockIdx.z;

    const float* xb = x + (size_t)b * CCH * HW;

    f32x8 acc[4][4];
#pragma unroll
    for (int it = 0; it < 4; ++it)
#pragma unroll
        for (int jt = 0; jt < 4; ++jt)
            acc[it][jt] = (f32x8){0.f,0.f,0.f,0.f,0.f,0.f,0.f,0.f};

    for (int kb = 0; kb < CCH; kb += 32) {
        bf16x16 a[4], bb[4];
#pragma unroll
        for (int it = 0; it < 4; ++it)
            a[it] = frag_f32_row(w + (size_t)(tmb * 64 + it * 16 + l15) * CCH + kb, lane);
#pragma unroll
        for (int jt = 0; jt < 4; ++jt)
            bb[jt] = frag_f32_col(xb + (tnb * 64 + jt * 16 + l15), HW, kb, lane);
#pragma unroll
        for (int it = 0; it < 4; ++it)
#pragma unroll
            for (int jt = 0; jt < 4; ++jt)
                acc[it][jt] = wmma_bf16(a[it], bb[jt], acc[it][jt]);
    }

    // windowed scatter (pack adjacent channel pairs -> merged b32 stores)
#pragma unroll
    for (int jt = 0; jt < 4; ++jt) {
        const int n   = tnb * 64 + jt * 16 + l15;          // pixel (N)
        const int h   = n >> 6, wq = n & 63;
        const int ws1 = h >> 3, hh = h & 7, ws2 = wq >> 3, ww = wq & 7;
        const int win = hh * 8 + ww, p = ws1 * 8 + ws2;
#pragma unroll
        for (int it = 0; it < 4; ++it) {
#pragma unroll
            for (int j = 0; j < 4; ++j) {
                const int   m    = tmb * 64 + it * 16 + 2 * j + hi * 8; // even channel
                const float f0   = acc[it][jt][2 * j]     + bias[m];
                const float f1   = acc[it][jt][2 * j + 1] + bias[m + 1];
                const int   part = m >> 8;                 // 0=q 1=k 2=v
                const int   head = (m >> 5) & 7;
                const int   d    = m & 31;
                __bf16* dst = qkvws + (size_t)part * PART +
                    ((((size_t)(b * NHEAD + head) * NWIN + win) * NPOS + p) * DHEAD + d);
                dst[0] = (__bf16)f0;
                dst[1] = (__bf16)f1;
            }
        }
    }
}

// ---------------------------------------------------------------------------
// Kernel 2: per-window attention.  grid = (64 windows, 8 heads, 16 batches),
// 1 wave/block.  S = QK^T (4x4 WMMA tiles), softmax*(1/16), O = P V (4x2 x 2k).
// ---------------------------------------------------------------------------
__global__ __launch_bounds__(32) void win_attn(const __bf16* __restrict__ qkvws,
                                               __bf16* __restrict__ ows) {
    __shared__ __align__(16) __bf16 qs[NPOS * DHEAD];   // [p][d]
    __shared__ __align__(16) __bf16 ks[NPOS * DHEAD];   // [p][d]
    __shared__ __align__(16) __bf16 vT[DHEAD * NPOS];   // [d][p]  (transposed)
    __shared__ __align__(16) __bf16 Ps[NPOS * NPOS];    // softmax(P), [q][k]
    __shared__ float Ss[16 * NPOS];                     // one 16-row S block

    const int lane = threadIdx.x;
    const int l15  = lane & 15;
    const int hi   = (lane >> 4) & 1;
    const int win  = blockIdx.x, head = blockIdx.y, b = blockIdx.z;

    const size_t  base = ((size_t)(b * NHEAD + head) * NWIN + win) * NPOS * DHEAD;
    const __bf16* qg = qkvws + base;
    const __bf16* kg = qkvws + PART + base;
    const __bf16* vg = qkvws + 2 * PART + base;

    // stage q,k (b128 copies) and v transposed
    {
        const uint4* q4 = (const uint4*)qg;
        const uint4* k4 = (const uint4*)kg;
        uint4* qd = (uint4*)qs;
        uint4* kd = (uint4*)ks;
        for (int i = lane; i < NPOS * DHEAD / 8; i += 32) { qd[i] = q4[i]; kd[i] = k4[i]; }
        for (int i = lane; i < NPOS * DHEAD; i += 32) {
            const int p = i >> 5, d = i & 31;
            vT[d * NPOS + p] = vg[i];
        }
    }
    __syncthreads();

    const float scale = 0.0625f;  // 1/sqrt(256)

    // ---- S = Q K^T, row-block at a time; softmax -> Ps ----
#pragma unroll
    for (int i = 0; i < 4; ++i) {
        bf16x16 aq = frag_bf16(qs + (i * 16 + l15) * DHEAD, lane);
        f32x8 s[4];
#pragma unroll
        for (int j = 0; j < 4; ++j) {
            f32x8 z = {0.f, 0.f, 0.f, 0.f, 0.f, 0.f, 0.f, 0.f};
            bf16x16 bk = frag_bf16(ks + (j * 16 + l15) * DHEAD, lane);
            s[j] = wmma_bf16(aq, bk, z);
        }
#pragma unroll
        for (int j = 0; j < 4; ++j)
#pragma unroll
            for (int r = 0; r < 8; ++r)
                Ss[(r + hi * 8) * NPOS + j * 16 + l15] = s[j][r];
        __syncthreads();

        if (lane < 16) {
            float row[NPOS];
            float mx = -3.0e38f;
#pragma unroll
            for (int c = 0; c < NPOS; ++c) {
                row[c] = Ss[lane * NPOS + c] * scale;
                mx = fmaxf(mx, row[c]);
            }
            float sum = 0.f;
#pragma unroll
            for (int c = 0; c < NPOS; ++c) { row[c] = __expf(row[c] - mx); sum += row[c]; }
            const float inv = 1.0f / sum;
#pragma unroll
            for (int c = 0; c < NPOS; ++c)
                Ps[(i * 16 + lane) * NPOS + c] = (__bf16)(row[c] * inv);
        }
        __syncthreads();
    }

    // ---- O = P V ----
    f32x8 o[4][2];
#pragma unroll
    for (int i = 0; i < 4; ++i)
#pragma unroll
        for (int j = 0; j < 2; ++j) o[i][j] = (f32x8){0.f,0.f,0.f,0.f,0.f,0.f,0.f,0.f};

#pragma unroll
    for (int kstep = 0; kstep < 2; ++kstep) {
#pragma unroll
        for (int i = 0; i < 4; ++i) {
            bf16x16 ap = frag_bf16(Ps + (i * 16 + l15) * NPOS + kstep * 32, lane);
#pragma unroll
            for (int j = 0; j < 2; ++j) {
                bf16x16 bv = frag_bf16(vT + (j * 16 + l15) * NPOS + kstep * 32, lane);
                o[i][j] = wmma_bf16(ap, bv, o[i][j]);
            }
        }
    }

    // store O in OUTPUT spatial order: (h,w) = (hh*8+ws1, ww*8+ws2), bf16 (B,C,H,W)
    const int hh = win >> 3, ww = win & 7;
#pragma unroll
    for (int i = 0; i < 4; ++i)
#pragma unroll
        for (int j = 0; j < 2; ++j)
#pragma unroll
            for (int r = 0; r < 8; ++r) {
                const int p   = i * 16 + r + hi * 8;
                const int d   = j * 16 + l15;
                const int ws1 = p >> 3, ws2 = p & 7;
                const int c   = head * DHEAD + d;
                const size_t off = (((size_t)b * CCH + c) * 64 + (hh * 8 + ws1)) * 64 +
                                   (ww * 8 + ws2);
                ows[off] = (__bf16)o[i][j][r];
            }
}

// ---------------------------------------------------------------------------
// Kernel 3: out = w_proj @ o + b_proj, fp32 result.
// Each wave computes a 64x64 output block (4x4 WMMA tiles).
// grid = (4 Mblocks, 64 Nblocks, 16 batches), 1 wave/block.
// ---------------------------------------------------------------------------
__global__ __launch_bounds__(32) void proj_gemm(const __bf16* __restrict__ ows,
                                                const float* __restrict__ w,
                                                const float* __restrict__ bias,
                                                float* __restrict__ out) {
    const int lane = threadIdx.x;
    const int l15  = lane & 15;
    const int hi   = (lane >> 4) & 1;
    const int tmb  = blockIdx.x;   // 0..3   (256/64)
    const int tnb  = blockIdx.y;   // 0..63  (4096/64)
    const int b    = blockIdx.z;

    const __bf16* ob = ows + (size_t)b * CCH * HW;

    f32x8 acc[4][4];
#pragma unroll
    for (int it = 0; it < 4; ++it)
#pragma unroll
        for (int jt = 0; jt < 4; ++jt)
            acc[it][jt] = (f32x8){0.f,0.f,0.f,0.f,0.f,0.f,0.f,0.f};

    for (int kb = 0; kb < CCH; kb += 32) {
        bf16x16 a[4], bb[4];
#pragma unroll
        for (int it = 0; it < 4; ++it)
            a[it] = frag_f32_row(w + (size_t)(tmb * 64 + it * 16 + l15) * CCH + kb, lane);
#pragma unroll
        for (int jt = 0; jt < 4; ++jt)
            bb[jt] = frag_bf16_col(ob + (tnb * 64 + jt * 16 + l15), HW, kb, lane);
#pragma unroll
        for (int it = 0; it < 4; ++it)
#pragma unroll
            for (int jt = 0; jt < 4; ++jt)
                acc[it][jt] = wmma_bf16(a[it], bb[jt], acc[it][jt]);
    }

#pragma unroll
    for (int jt = 0; jt < 4; ++jt) {
        const int n = tnb * 64 + jt * 16 + l15;
#pragma unroll
        for (int it = 0; it < 4; ++it)
#pragma unroll
            for (int r = 0; r < 8; ++r) {
                const int m = tmb * 64 + it * 16 + r + hi * 8;
                out[((size_t)b * CCH + m) * HW + n] = acc[it][jt][r] + bias[m];
            }
    }
}

// ---------------------------------------------------------------------------
extern "C" void kernel_launch(void* const* d_in, const int* in_sizes, int n_in,
                              void* d_out, int out_size, void* d_ws, size_t ws_size,
                              hipStream_t stream) {
    const float* x      = (const float*)d_in[0];
    const float* w_qkv  = (const float*)d_in[1];
    const float* b_qkv  = (const float*)d_in[2];
    const float* w_proj = (const float*)d_in[3];
    const float* b_proj = (const float*)d_in[4];
    float* out = (float*)d_out;

    __bf16* ws    = (__bf16*)d_ws;
    __bf16* qkvws = ws;              // 3*PART bf16 (q,k,v windowed)
    __bf16* ows   = ws + 3 * PART;   // PART bf16 (attention out, BCHW)

    qkv_gemm<<<dim3(12, 64, NBATCH), 32, 0, stream>>>(x, w_qkv, b_qkv, qkvws);
    win_attn<<<dim3(NWIN, NHEAD, NBATCH), 32, 0, stream>>>(qkvws, ows);
    proj_gemm<<<dim3(4, 64, NBATCH), 32, 0, stream>>>(ows, w_proj, b_proj, out);
}
